// KnotForward_86827058856331
// MI455X (gfx1250) — compile-verified, matrix operand order checked by reading, not run on hardware
//
#include <hip/hip_runtime.h>
#include <hip/hip_bf16.h>

#define MAXK 64
#define HID 64
#define DIN 3
#define HS 67          // HID + DIN
#define KPAD 96        // HS padded to 3 k-chunks of 32
#define NTHR 256
#define NCAND 4096     // MAXK + (MAXK-1)*HID
#define EPSV 1e-6f
#define BIGV 2.0f

typedef _Float16 v16h __attribute__((ext_vector_type(16)));
typedef _Float16 v8h  __attribute__((ext_vector_type(8)));
typedef float    v8f  __attribute__((ext_vector_type(8)));

// Load a 16-half WMMA fragment from LDS, ISA 7.12.2 16-bit layout:
// lanes 0-15 take K = base+0..7 and base+16..23 ; lanes 16-31 take +8 / +24.
__device__ __forceinline__ v16h load_frag(const _Float16* base, int row, int col0, int stride) {
  const v8h lo = *reinterpret_cast<const v8h*>(base + row * stride + col0);
  const v8h hi = *reinterpret_cast<const v8h*>(base + row * stride + col0 + 16);
  v16h r;
#pragma unroll
  for (int i = 0; i < 8; i++) { r[i] = lo[i]; r[8 + i] = hi[i]; }
  return r;
}

__device__ __forceinline__ void cmp_xchg(volatile unsigned long long* keys,
                                         unsigned i, unsigned j, unsigned k) {
  unsigned p = i | j;                       // i's j-bit is 0 by construction
  unsigned long long a = keys[i], b = keys[p];
  if ((a > b) == ((i & k) == 0u)) { keys[i] = b; keys[p] = a; }
}

// Hierarchical bitonic sort of 4096 u64 keys, ascending.
// Each wave owns a contiguous 512-key chunk; every step with j<512 stays inside
// one chunk (i and i^j share i&~511), so it needs no workgroup barrier:
// per-wave LDS ops are in-order (DScnt), volatile + wave_barrier pins the
// compiler. Only the 6 cross-chunk merge steps (j>=512) use s_barrier.
__device__ void sort_keys(unsigned long long* keys_, int tid) {
  volatile unsigned long long* keys = keys_;
  const unsigned lane  = (unsigned)(tid & 31);
  const unsigned chunk = ((unsigned)(tid >> 5)) << 9;   // wave * 512

  // local phase: k = 2..512 entirely within each wave's chunk
  for (unsigned k = 2; k <= 512u; k <<= 1) {
    for (unsigned j = k >> 1; j > 0; j >>= 1) {
      for (unsigned q = lane; q < 256u; q += 32u) {
        unsigned il = ((q & ~(j - 1u)) << 1) | (q & (j - 1u));
        cmp_xchg(keys, chunk + il, j, k);
      }
      __builtin_amdgcn_wave_barrier();
    }
  }
  __syncthreads();

  // merge phases k = 1024, 2048, 4096
  for (unsigned k = 1024; k <= 4096u; k <<= 1) {
    for (unsigned j = k >> 1; j >= 512u; j >>= 1) {      // cross-chunk steps
      for (unsigned q = (unsigned)tid; q < 2048u; q += NTHR) {
        unsigned i = ((q & ~(j - 1u)) << 1) | (q & (j - 1u));
        cmp_xchg(keys, i, j, k);
      }
      __syncthreads();
    }
    for (unsigned j = 256u; j > 0; j >>= 1) {            // within-chunk steps
      for (unsigned q = lane; q < 256u; q += 32u) {
        unsigned il = ((q & ~(j - 1u)) << 1) | (q & (j - 1u));
        cmp_xchg(keys, chunk + il, j, k);
      }
      __builtin_amdgcn_wave_barrier();
    }
    __syncthreads();
  }
}

// Top-64-by-t knot insertion: candidate gen -> stable sort (keys in LDS)
// -> decode/interp into registers -> write back over zbuf/tk/vk.
__device__ void insert_crossings(float (*zbuf)[HID + 1], float* tk, int* vk,
                                 unsigned long long* keys, int tid) {
  // ---- candidate generation: 64 originals + 4032 crossings ----
  for (int i = tid; i < NCAND; i += NTHR) {
    float tv;
    if (i < MAXK) {
      tv = vk[i] ? tk[i] : BIGV;
    } else {
      int m   = i - MAXK;
      int seg = m >> 6;       // / HID
      int hh  = m & 63;       // % HID
      float zl = zbuf[seg][hh], zr = zbuf[seg + 1][hh];
      float den = zr - zl;
      tv = BIGV;
      if (vk[seg] && vk[seg + 1] && fabsf(den) > EPSV) {
        float al = -zl / den;
        if (al > EPSV && al < 1.0f - EPSV)
          tv = tk[seg] + al * (tk[seg + 1] - tk[seg]);
      }
    }
    // t >= 0 so the float bit pattern is order-preserving; low 32 bits give a
    // stable tie-break identical to jnp.argsort over the concatenation order.
    keys[i] = (((unsigned long long)__float_as_uint(tv)) << 32) | (unsigned)i;
  }
  __syncthreads();

  sort_keys(keys, tid);

  // ---- decode first 64 keys; interpolate z; stage in registers ----
  float znew[16], tnew[16];
  int   vnew[16];
#pragma unroll
  for (int s = 0; s < 16; s++) {
    int e = tid * 16 + s;          // element = knot*64 + col
    int knot = e >> 6, col = e & 63;
    unsigned long long key = keys[knot];
    float tkey  = __uint_as_float((unsigned)(key >> 32));
    unsigned idx = (unsigned)(key & 0xffffffffu);
    bool vv = tkey < BIGV;
    float zv = 0.0f;
    if (vv) {
      if (idx < MAXK) {
        zv = zbuf[idx][col];
      } else {
        int m = (int)idx - MAXK;
        int seg = m >> 6, hh = m & 63;
        float zl = zbuf[seg][hh], zr = zbuf[seg + 1][hh];
        float al = -zl / (zr - zl);
        zv = zbuf[seg][col] + al * (zbuf[seg + 1][col] - zbuf[seg][col]);
      }
    }
    znew[s] = zv;
    tnew[s] = vv ? tkey : 1.0f;
    vnew[s] = vv ? 1 : 0;
  }
  __syncthreads();                  // all reads of old zbuf/keys done
#pragma unroll
  for (int s = 0; s < 16; s++) {
    int e = tid * 16 + s;
    int knot = e >> 6, col = e & 63;
    zbuf[knot][col] = znew[s];
    if (col == 0) { tk[knot] = tnew[s]; vk[knot] = vnew[s]; }
  }
  __syncthreads();
}

// One hidden layer: h_input = [relu(z) | coords] (f16, padded to KPAD),
// z = h_input @ W^T + b via v_wmma_f32_16x16x32_f16, masked by validity.
__device__ void layer_matmul(float (*zbuf)[HID + 1], const float* tk, const int* vk,
                             _Float16* aF, _Float16* wF,
                             const float* W, const float* bvec,
                             float s0, float s1, float s2,
                             float d0, float d1, float d2, int tid) {
  // stage A (activations+coords) and B (weights) as f16 in LDS
  for (int e = tid; e < MAXK * KPAD; e += NTHR) {
    int k = e / KPAD, c = e % KPAD;
    float val = 0.0f;
    if (vk[k]) {
      if (c < HID)           val = fmaxf(zbuf[k][c], 0.0f);
      else if (c == HID)     val = s0 + tk[k] * d0;
      else if (c == HID + 1) val = s1 + tk[k] * d1;
      else if (c == HID + 2) val = s2 + tk[k] * d2;
    }
    aF[e] = (_Float16)val;
  }
  for (int e = tid; e < HID * KPAD; e += NTHR) {
    int h = e / KPAD, f = e % KPAD;
    wF[e] = (f < HS) ? (_Float16)W[h * HS + f] : (_Float16)0.0f;
  }
  __syncthreads();

  int lane = tid & 31;
  int wave = tid >> 5;
  int mt  = wave >> 1;             // output row tile 0..3
  int nt0 = (wave & 1) * 2;        // first of two col tiles
  int rowA  = mt * 16 + (lane & 15);
  int rowB0 = nt0 * 16 + (lane & 15);
  int rowB1 = rowB0 + 16;
  int koff  = (lane >> 4) * 8;

  v8f c0 = {}; v8f c1 = {};
#pragma unroll
  for (int kc = 0; kc < 3; kc++) {
    int col0 = kc * 32 + koff;
    v16h a   = load_frag(aF, rowA,  col0, KPAD);
    v16h bb0 = load_frag(wF, rowB0, col0, KPAD);
    v16h bb1 = load_frag(wF, rowB1, col0, KPAD);
    c0 = __builtin_amdgcn_wmma_f32_16x16x32_f16(false, a, false, bb0, (short)0, c0, false, false);
    c1 = __builtin_amdgcn_wmma_f32_16x16x32_f16(false, a, false, bb1, (short)0, c1, false, false);
  }
  __syncthreads();                  // everyone done reading aF/wF region

  int half = lane >> 4;
  int nloc = lane & 15;
#pragma unroll
  for (int r = 0; r < 8; r++) {     // C layout: VGPR r -> M = r + 8*half
    int m  = mt * 16 + r + half * 8;
    int h0 = nt0 * 16 + nloc;
    int h1 = h0 + 16;
    float z0 = c0[r] + bvec[h0];
    float z1 = c1[r] + bvec[h1];
    if (!vk[m]) { z0 = 0.0f; z1 = 0.0f; }
    zbuf[m][h0] = z0;
    zbuf[m][h1] = z1;
  }
  __syncthreads();
}

__global__ __launch_bounds__(NTHR)
void KnotForward_86827058856331_kernel(const float* __restrict__ end_p,
                                       const float* __restrict__ start_p,
                                       const float* __restrict__ W0, const float* __restrict__ b0,
                                       const float* __restrict__ W1, const float* __restrict__ b1,
                                       const float* __restrict__ W2, const float* __restrict__ b2,
                                       const float* __restrict__ W3, const float* __restrict__ b3,
                                       float* __restrict__ t_out,
                                       float* __restrict__ v_out,
                                       float* __restrict__ y_out, int B) {
  __shared__ float zbuf[MAXK][HID + 1];                 // 64x65 f32, padded stride
  __shared__ float tk[MAXK];
  __shared__ int   vk[MAXK];
  __shared__ float a0s[HID], b0s[HID];
  __shared__ __align__(16) unsigned long long scratch64[NCAND];  // 32KB: keys / f16 staging

  unsigned long long* keys = scratch64;
  _Float16* aF = reinterpret_cast<_Float16*>(scratch64);          // 64*96 halves = 12KB
  _Float16* wF = aF + MAXK * KPAD;                                // next 12KB

  const int blk = blockIdx.x;
  const int tid = threadIdx.x;
  if (blk >= B) return;

  const float s0 = start_p[blk * 3 + 0], s1 = start_p[blk * 3 + 1], s2 = start_p[blk * 3 + 2];
  const float d0 = end_p[blk * 3 + 0] - s0, d1 = end_p[blk * 3 + 1] - s1, d2 = end_p[blk * 3 + 2] - s2;

  if (tid < MAXK) {
    tk[tid] = (tid == 0) ? 0.0f : 1.0f;
    vk[tid] = (tid < 2) ? 1 : 0;
  }
  if (tid < HID) {                       // alpha0 = d@W0^T ; beta0 = start@W0^T + b0
    float a = 0.0f, b = b0[tid];
    a += d0 * W0[tid * 3 + 0] + d1 * W0[tid * 3 + 1] + d2 * W0[tid * 3 + 2];
    b += s0 * W0[tid * 3 + 0] + s1 * W0[tid * 3 + 1] + s2 * W0[tid * 3 + 2];
    a0s[tid] = a; b0s[tid] = b;
  }
  __syncthreads();

  // layer-0 pre-activations: z[k][h] = a0[h]*t[k] + b0[h], masked
  for (int e = tid; e < MAXK * HID; e += NTHR) {
    int k = e >> 6, h = e & 63;
    zbuf[k][h] = vk[k] ? fmaf(a0s[h], tk[k], b0s[h]) : 0.0f;
  }
  __syncthreads();

  insert_crossings(zbuf, tk, vk, keys, tid);

  layer_matmul(zbuf, tk, vk, aF, wF, W1, b1, s0, s1, s2, d0, d1, d2, tid);
  insert_crossings(zbuf, tk, vk, keys, tid);

  layer_matmul(zbuf, tk, vk, aF, wF, W2, b2, s0, s1, s2, d0, d1, d2, tid);
  insert_crossings(zbuf, tk, vk, keys, tid);

  // output head: y[k] = relu(z[k]) . W3 + b3, masked
  if (tid < MAXK) {
    float acc = b3[0];
#pragma unroll 8
    for (int h = 0; h < HID; h++) acc += fmaxf(zbuf[tid][h], 0.0f) * W3[h];
    const int o = blk * MAXK + tid;
    t_out[o] = tk[tid];
    v_out[o] = vk[tid] ? 1.0f : 0.0f;
    y_out[o] = vk[tid] ? acc : 0.0f;
  }
}

extern "C" void kernel_launch(void* const* d_in, const int* in_sizes, int n_in,
                              void* d_out, int out_size, void* d_ws, size_t ws_size,
                              hipStream_t stream) {
  const float* end_p   = (const float*)d_in[0];
  const float* start_p = (const float*)d_in[1];
  const float* W0 = (const float*)d_in[2];
  const float* b0 = (const float*)d_in[3];
  const float* W1 = (const float*)d_in[4];
  const float* b1 = (const float*)d_in[5];
  const float* W2 = (const float*)d_in[6];
  const float* b2 = (const float*)d_in[7];
  const float* W3 = (const float*)d_in[8];
  const float* b3 = (const float*)d_in[9];

  const int B = in_sizes[0] / 3;          // end_points is (B,3)
  float* out = (float*)d_out;
  float* t_out = out;                     // (B,64)
  float* v_out = out + (size_t)B * MAXK;  // (B,64) valid as 0/1
  float* y_out = out + 2 * (size_t)B * MAXK;  // (B,64,1)

  KnotForward_86827058856331_kernel<<<B, NTHR, 0, stream>>>(
      end_p, start_p, W0, b0, W1, b1, W2, b2, W3, b3, t_out, v_out, y_out, B);
}
